// Encoder_60017872995167
// MI455X (gfx1250) — matrix operand order repaired and checked by reading, not compiled, not run on hardware
//
#include <hip/hip_runtime.h>
#include <hip/hip_bf16.h>

typedef __attribute__((ext_vector_type(16))) __bf16 v16bf;
typedef __attribute__((ext_vector_type(8)))  float  v8f;

#define B_  32
#define T_  1024
#define E_  256
#define H_  512
#define H3_ 1536
#define ROWS (B_ * T_)          // 32768
#define NSCAN_BLOCKS 16
#define EPS_ 1e-3f

// ---------------------------------------------------------------- swizzle maps
// A-fragment layout (16x32 bf16 tile): lane = h*16 + (m%16), h = K-octet half.
// Lane's 16 contiguous values: i<8 -> K = h*8 + i ; i>=8 -> K = h*8 + 16 + (i-8).
__device__ __forceinline__ size_t a_swz(int m, int k, int Kt) {
    int mt = m >> 4, mr = m & 15;
    int kt = k >> 5, r = k & 31;
    int h = (r >> 3) & 1, c = r >> 4, p = r & 7;
    return ((size_t)(mt * Kt + kt) * 32 + h * 16 + mr) * 16 + c * 8 + p;
}
// B-fragment layout (32x16 bf16 tile): lane = h*16 + (n%16); values i<8 -> K = h*8 + i ; i>=8 -> K = h*8 + 16 + (i-8).
__device__ __forceinline__ size_t b_swz(int k, int n, int Kt) {
    int nt = n >> 4, col = n & 15;
    int kt = k >> 5, r = k & 31;
    int h = (r >> 3) & 1, i = (r >> 4) * 8 + (r & 7);
    return ((size_t)(nt * Kt + kt) * 32 + h * 16 + col) * 16 + i;
}

__device__ __forceinline__ float sigmoidf_(float x) {
    return 1.0f / (1.0f + __expf(-x));
}

__device__ __forceinline__ void grid_barrier(unsigned* bar, int nblocks) {
    __syncthreads();
    if (threadIdx.x == 0) {
        volatile unsigned* vbar = (volatile unsigned*)bar;
        unsigned gen = vbar[1];
        __threadfence();
        if (atomicAdd(&bar[0], 1u) == (unsigned)(nblocks - 1)) {
            vbar[0] = 0u;
            __threadfence();
            atomicAdd(&bar[1], 1u);
        } else {
            while (vbar[1] == gen) { __builtin_amdgcn_s_sleep(2); }
        }
    }
    __syncthreads();
}

// ------------------------------------------------------------- producers
// Convert f32 weight [K x N] (row-major) -> bf16 in B-fragment-swizzled layout.
__global__ void cvt_bswz(const float* __restrict__ src, __bf16* __restrict__ dst,
                         int K, int N) {
    int i = blockIdx.x * blockDim.x + threadIdx.x;
    if (i < K * N) {
        int k = i / N, n = i - k * N;
        dst[b_swz(k, n, K >> 5)] = (__bf16)src[i];
    }
}

// Embedding lookup -> A-swizzled bf16 [ROWS x E_], Kt = 8.
__global__ void embed_kernel(const int* __restrict__ x, const float* __restrict__ emb,
                             __bf16* __restrict__ out) {
    int row = blockIdx.x;            // 0..32767  (b*T + t)
    int e   = threadIdx.x;           // 0..255
    int id  = x[row];
    out[a_swz(row, e, E_ >> 5)] = (__bf16)emb[(size_t)id * E_ + e];
}

__global__ void init_kernel(float* __restrict__ hF, __bf16* __restrict__ hBf,
                            unsigned* __restrict__ bar) {
    int i = blockIdx.x * blockDim.x + threadIdx.x;
    if (i < B_ * H_) { hF[i] = 0.0f; hBf[i] = (__bf16)0.0f; }
    if (i < 2) bar[i] = 0u;
}

__global__ void copy_state(const float* __restrict__ hF, float* __restrict__ dst) {
    int i = blockIdx.x * blockDim.x + threadIdx.x;
    if (i < B_ * H_) dst[i] = hF[i];
}

// ------------------------------------------------------- xproj WMMA GEMM
// Y[32768 x 1536] = A_swz * B_swz + bias. No LDS: fragments are contiguous
// 32B/lane global_load_b128 pairs from the swizzled operands (L2-resident).
// Block 256 thr (8 waves, 4x2); wave computes 32x32 via 2x2 wmma tiles.
// K-loop is software-pipelined with a register double buffer so the next
// tile's loads overlap the current tile's WMMAs.
__global__ __launch_bounds__(256) void gemm_xproj(
        const __bf16* __restrict__ A, int Kt,
        const __bf16* __restrict__ Bw, const float* __restrict__ bias,
        float* __restrict__ Y) {
    const int tid  = threadIdx.x;
    const int lane = tid & 31;
    const int wv   = tid >> 5;
    const int m0   = blockIdx.y * 128;
    const int n0   = blockIdx.x * 64;
    const int wm   = (wv & 3) * 32;
    const int wn   = (wv >> 2) * 32;
    const int lr   = lane & 15;
    const int hi   = lane >> 4;
    const int mtIdx = (m0 + wm) >> 4;   // global 16-row tile index
    const int ntIdx = (n0 + wn) >> 4;   // global 16-col tile index

    v8f acc[2][2] = {};
    const size_t laneOff = (size_t)lane * 16;

    const __bf16* aBase = A  + laneOff;
    const __bf16* bBase = Bw + laneOff;

    v16bf af0[2], bf0[2], af1[2], bf1[2];

    // prologue: fetch kt = 0
    #pragma unroll
    for (int mt = 0; mt < 2; ++mt)
        af0[mt] = *(const v16bf*)&aBase[((size_t)(mtIdx + mt) * Kt) * 512];
    #pragma unroll
    for (int nt = 0; nt < 2; ++nt)
        bf0[nt] = *(const v16bf*)&bBase[((size_t)(ntIdx + nt) * Kt) * 512];

    for (int kt = 0; kt < Kt; kt += 2) {            // Kt is even (8 or 16)
        // fetch kt+1 into buffer 1
        #pragma unroll
        for (int mt = 0; mt < 2; ++mt)
            af1[mt] = *(const v16bf*)&aBase[((size_t)(mtIdx + mt) * Kt + kt + 1) * 512];
        #pragma unroll
        for (int nt = 0; nt < 2; ++nt)
            bf1[nt] = *(const v16bf*)&bBase[((size_t)(ntIdx + nt) * Kt + kt + 1) * 512];
        __builtin_prefetch(&aBase[((size_t)mtIdx * Kt + kt + 2) * 512], 0, 1);
        // compute kt
        #pragma unroll
        for (int mt = 0; mt < 2; ++mt)
            #pragma unroll
            for (int nt = 0; nt < 2; ++nt)
                acc[mt][nt] = __builtin_amdgcn_wmma_f32_16x16x32_bf16(
                    false, af0[mt], false, bf0[nt], (short)0, acc[mt][nt], false, false);
        // fetch kt+2 into buffer 0
        if (kt + 2 < Kt) {
            #pragma unroll
            for (int mt = 0; mt < 2; ++mt)
                af0[mt] = *(const v16bf*)&aBase[((size_t)(mtIdx + mt) * Kt + kt + 2) * 512];
            #pragma unroll
            for (int nt = 0; nt < 2; ++nt)
                bf0[nt] = *(const v16bf*)&bBase[((size_t)(ntIdx + nt) * Kt + kt + 2) * 512];
        }
        // compute kt+1
        #pragma unroll
        for (int mt = 0; mt < 2; ++mt)
            #pragma unroll
            for (int nt = 0; nt < 2; ++nt)
                acc[mt][nt] = __builtin_amdgcn_wmma_f32_16x16x32_bf16(
                    false, af1[mt], false, bf1[nt], (short)0, acc[mt][nt], false, false);
    }

    // C layout: N = lane&15; VGPR e -> M = e + (lane>=16)*8
    #pragma unroll
    for (int mt = 0; mt < 2; ++mt)
        #pragma unroll
        for (int nt = 0; nt < 2; ++nt) {
            int n = n0 + wn + nt * 16 + lr;
            float bv = bias[n];
            #pragma unroll
            for (int e = 0; e < 8; ++e) {
                int m = m0 + wm + mt * 16 + e + hi * 8;
                Y[(size_t)m * H3_ + n] = acc[mt][nt][e] + bv;
            }
        }
}

// -------------------------------------------------- persistent GRU scan
// 16 blocks x 384 threads (12 waves). Block owns hidden channels [jb, jb+32).
// Wave owns one 16x16 tile of inner = h @ rkernel; its 512x16 weight slice is
// 16 v16bf fragments held in VGPRs across the whole time loop (from B-swizzled
// global). h (bf16) lives in global in A-swizzled order; staged to LDS flat.
__global__ __launch_bounds__(384, 1) void gru_scan(
        const float* __restrict__ xproj, const __bf16* __restrict__ rkw,
        const float* __restrict__ br, float* __restrict__ hF,
        __bf16* __restrict__ hBf, float* __restrict__ yF,
        unsigned* __restrict__ bar) {
    __shared__ __attribute__((aligned(32))) __bf16 hS[B_ * H_];   // 32 KB, swizzled
    __shared__ float innerS[32][100];

    const int tid   = threadIdx.x;
    const int lane  = tid & 31;
    const int wv    = tid >> 5;       // 0..11
    const int jb    = blockIdx.x * 32;
    const int mtile = wv & 1;         // rows 0-15 / 16-31
    const int ntile = wv >> 1;        // 0..5 over 96 gate columns
    const int g     = ntile >> 1;     // gate 0=z 1=r 2=h
    const int lr    = lane & 15;
    const int hi    = lane >> 4;
    // global 16-col tile index of this wave's slice within [K=512 x N=1536] rkernel
    const int ntg   = (g * H_ + jb + (ntile & 1) * 16) >> 4;
    const size_t laneOff = (size_t)lane * 16;

    // Preload recurrent-weight fragments (vectorized 32B/lane loads, L2-hot).
    v16bf bfrag[16];
    #pragma unroll
    for (int kk = 0; kk < 16; ++kk)
        bfrag[kk] = *(const v16bf*)&rkw[((size_t)ntg * 16 + kk) * 512 + laneOff];

    for (int t = 0; t < T_; ++t) {
        // Stage swizzled h into LDS: 16384 bf16 = 2048 x 16B, coalesced.
        for (int c = tid; c < (B_ * H_) / 8; c += 384)
            ((uint4*)hS)[c] = ((const uint4*)hBf)[c];
        __syncthreads();

        v8f acc = {};
        #pragma unroll
        for (int kk = 0; kk < 16; ++kk) {
            v16bf a = *(const v16bf*)&hS[((size_t)(mtile * 16 + kk) * 32) * 16 + laneOff];
            acc = __builtin_amdgcn_wmma_f32_16x16x32_bf16(
                false, a, false, bfrag[kk], (short)0, acc, false, false);
        }

        // Scatter inner tile to LDS (C layout)
        {
            int n  = ntile * 16 + lr;
            int mb = mtile * 16 + hi * 8;
            #pragma unroll
            for (int e = 0; e < 8; ++e) innerS[mb + e][n] = acc[e];
        }
        __syncthreads();

        // Gate math for this block's 32 channels x 32 batches
        for (int idx = tid; idx < B_ * 32; idx += 384) {
            int c = idx & 31, b = idx >> 5;
            int j = jb + c;
            float hz = innerS[b][c]      + br[j];
            float hr = innerS[b][32 + c] + br[H_ + j];
            float hh = innerS[b][64 + c] + br[2 * H_ + j];
            size_t row = (size_t)b * T_ + t;
            float xz = xproj[row * H3_ + j];
            float xr = xproj[row * H3_ + H_ + j];
            float xh = xproj[row * H3_ + 2 * H_ + j];
            float z    = sigmoidf_(xz + hz);
            float rr   = sigmoidf_(xr + hr);
            float cand = xh + rr * hh;
            cand = cand > 0.0f ? cand : 0.0f;              // relu
            float hold = hF[b * H_ + j];
            float hn   = z * hold + (1.0f - z) * cand;
            hF[b * H_ + j]            = hn;
            hBf[a_swz(b, j, H_ >> 5)] = (__bf16)hn;        // A-swizzled for next step
            yF[row * H_ + j]          = hn;
        }
        grid_barrier(bar, NSCAN_BLOCKS);
    }
}

// ----------------------------------------------------------- batch norm
__global__ __launch_bounds__(256) void bn_kernel(
        const float* __restrict__ y, const float* __restrict__ gamma,
        const float* __restrict__ beta, float* __restrict__ outF,
        __bf16* __restrict__ outBf) {
    __shared__ float s1[256];
    __shared__ float s2[256];
    const int ch = blockIdx.x, tid = threadIdx.x;
    float sum = 0.0f, sq = 0.0f;
    for (int i = tid; i < ROWS; i += 256) {
        float v = y[(size_t)i * H_ + ch];
        sum += v; sq += v * v;
    }
    s1[tid] = sum; s2[tid] = sq;
    __syncthreads();
    for (int s = 128; s > 0; s >>= 1) {
        if (tid < s) { s1[tid] += s1[tid + s]; s2[tid] += s2[tid + s]; }
        __syncthreads();
    }
    float mean  = s1[0] * (1.0f / ROWS);
    float var   = s2[0] * (1.0f / ROWS) - mean * mean;
    float scale = gamma[ch] * rsqrtf(var + EPS_);
    float shift = beta[ch] - mean * scale;
    for (int i = tid; i < ROWS; i += 256) {
        float v = y[(size_t)i * H_ + ch] * scale + shift;
        if (outF)  outF[(size_t)i * H_ + ch] = v;
        if (outBf) outBf[a_swz(i, ch, H_ >> 5)] = (__bf16)v;   // next layer A operand
    }
}

// ---------------------------------------------------------------- launch
extern "C" void kernel_launch(void* const* d_in, const int* in_sizes, int n_in,
                              void* d_out, int out_size, void* d_ws, size_t ws_size,
                              hipStream_t stream) {
    const int*   x      = (const int*)  d_in[0];
    const float* emb    = (const float*)d_in[1];
    const float* k0     = (const float*)d_in[2];
    const float* rk0    = (const float*)d_in[3];
    const float* b0     = (const float*)d_in[4];
    const float* krest  = (const float*)d_in[5];
    const float* rkrest = (const float*)d_in[6];
    const float* brest  = (const float*)d_in[7];
    const float* gammas = (const float*)d_in[8];
    const float* betas  = (const float*)d_in[9];
    float* out = (float*)d_out;

    char* w = (char*)d_ws;
    auto take = [&](size_t bytes) -> char* {
        char* p = w;
        w += (bytes + 255) & ~(size_t)255;
        return p;
    };
    float*   xprojF = (float*)  take((size_t)ROWS * H3_ * 4);
    __bf16*  XinBf  = (__bf16*) take((size_t)ROWS * H_ * 2);
    float*   yF     = (float*)  take((size_t)ROWS * H_ * 4);
    __bf16*  kB[3];
    kB[0] = (__bf16*)take((size_t)E_ * H3_ * 2);
    kB[1] = (__bf16*)take((size_t)H_ * H3_ * 2);
    kB[2] = (__bf16*)take((size_t)H_ * H3_ * 2);
    __bf16*  rkB[3];
    rkB[0] = (__bf16*)take((size_t)H_ * H3_ * 2);
    rkB[1] = (__bf16*)take((size_t)H_ * H3_ * 2);
    rkB[2] = (__bf16*)take((size_t)H_ * H3_ * 2);
    float*    hF  = (float*)   take((size_t)B_ * H_ * 4);
    __bf16*   hBf = (__bf16*)  take((size_t)B_ * H_ * 2);
    unsigned* bar = (unsigned*)take(256);

    const int nk0 = E_ * H3_;   // 393216
    const int nkr = H_ * H3_;   // 786432

    cvt_bswz<<<(nk0 + 255) / 256, 256, 0, stream>>>(k0,           kB[0],  E_, H3_);
    cvt_bswz<<<(nkr + 255) / 256, 256, 0, stream>>>(krest,        kB[1],  H_, H3_);
    cvt_bswz<<<(nkr + 255) / 256, 256, 0, stream>>>(krest + nkr,  kB[2],  H_, H3_);
    cvt_bswz<<<(nkr + 255) / 256, 256, 0, stream>>>(rk0,          rkB[0], H_, H3_);
    cvt_bswz<<<(nkr + 255) / 256, 256, 0, stream>>>(rkrest,       rkB[1], H_, H3_);
    cvt_bswz<<<(nkr + 255) / 256, 256, 0, stream>>>(rkrest + nkr, rkB[2], H_, H3_);

    embed_kernel<<<ROWS, 256, 0, stream>>>(x, emb, XinBf);
    init_kernel<<<64, 256, 0, stream>>>(hF, hBf, bar);

    const float* bi[3]  = { b0,        brest,        brest + 2 * H3_ };
    const float* brr[3] = { b0 + H3_,  brest + H3_,  brest + 3 * H3_ };

    for (int L = 0; L < 3; ++L) {
        int Kt = ((L == 0) ? E_ : H_) >> 5;     // K tiles of 32
        gemm_xproj<<<dim3(H3_ / 64, ROWS / 128), 256, 0, stream>>>(
            XinBf, Kt, kB[L], bi[L], xprojF);
        gru_scan<<<NSCAN_BLOCKS, 384, 0, stream>>>(
            xprojF, rkB[L], brr[L], hF, hBf, yF, bar);
        bn_kernel<<<H_, 256, 0, stream>>>(
            yF, gammas + L * H_, betas + L * H_,
            (L == 2) ? out : (float*)nullptr,
            (L < 2) ? XinBf : (__bf16*)nullptr);
    }
    copy_state<<<64, 256, 0, stream>>>(hF, out + (size_t)ROWS * H_);
}
